// CouplingFlow_1340029797001
// MI455X (gfx1250) — matrix operand order repaired and checked by reading, not compiled, not run on hardware
//
#include <hip/hip_runtime.h>
#include <math.h>

typedef __attribute__((ext_vector_type(16))) _Float16 v16h;
typedef __attribute__((ext_vector_type(8)))  float    v8f;

#define WAVES 8
#define TILES 4
#define BLOCK 256
#define SAMPLES_PER_BLOCK (WAVES * TILES * 16)   // 512
#define NLAYER 16
#define NEG_SLOPE 0.01f

union HPack { unsigned u; _Float16 h[2]; };
union AMat  { v16h v; unsigned u[8]; _Float16 e[16]; };

__device__ __forceinline__ float leaky(float x) { return fmaxf(x, NEG_SLOPE * x); }

// ---------------------------------------------------------------------------
// Kernel 1: full 16-layer coupling flow. Each wave carries TILES*16 samples
// in registers. Layer-2 (32x32) matmuls use v_wmma_f32_16x16x32_f16 with
// A = W2^T (weights, staged in LDS in the A-matrix f16 layout) and
// B = activations (lane = sample), so C keeps N = lane&15 = sample.
// ---------------------------------------------------------------------------
__global__ __launch_bounds__(BLOCK) void flow_kernel(
    const float* __restrict__ x,
    const float* __restrict__ sW1, const float* __restrict__ sb1,
    const float* __restrict__ sW2, const float* __restrict__ sb2,
    const float* __restrict__ sW3, const float* __restrict__ sb3,
    const float* __restrict__ tW1, const float* __restrict__ tb1,
    const float* __restrict__ tW2, const float* __restrict__ tb2,
    const float* __restrict__ tW3, const float* __restrict__ tb3,
    const float* __restrict__ scale, const float* __restrict__ mask,
    float* __restrict__ out, float* __restrict__ partials, int N)
{
    __shared__ unsigned aPack[2][2][256];   // [mlp][outTile][lane*8+v], packed f16 pairs (W2^T in A-layout)
    __shared__ float2   pk1[2][32];         // (W1[active,k], b1[k])
    __shared__ float    b2l[2][32];
    __shared__ float2   pw3[2][32];         // (W3[j,0], W3[j,1])
    __shared__ float    b3l[2][2];
    __shared__ float    scl[2];
    __shared__ float    redbuf[WAVES];

    const int tid  = threadIdx.x;
    const int lane = tid & 31;
    const int wave = tid >> 5;
    const int hi   = lane >> 4;      // which 16-lane half
    const int m    = lane & 15;      // sample within tile

    // x state: lane (and its partner lane^16) holds sample base + t*16 + m
    float xs0[TILES], xs1[TILES];
    bool  valid[TILES];
    const int base = blockIdx.x * SAMPLES_PER_BLOCK + wave * (TILES * 16);
    #pragma unroll
    for (int t = 0; t < TILES; ++t) {
        int n = base + t * 16 + m;
        valid[t] = (n < N);
        xs0[t] = valid[t] ? x[2 * n]     : 0.0f;
        xs1[t] = valid[t] ? x[2 * n + 1] : 0.0f;
    }
    float ldj = 0.0f;

    for (int l = 0; l < NLAYER; ++l) {
        const int a = (mask[2 * l] > 0.5f) ? 0 : 1;   // active (kept) coordinate

        __syncthreads();   // previous layer done reading LDS
        // ---- stage A = W2^T in f16 A-matrix layout (both MLPs, both 16-col output tiles)
        for (int idx = tid; idx < 1024; idx += BLOCK) {
            int mlp  = idx >> 9;
            int tile = (idx >> 8) & 1;
            int e    = idx & 255;
            int el   = e & 31;
            int v    = e >> 5;
            int ehi  = el >> 4;
            int em   = el & 15;
            int nOut = tile * 16 + em;                       // A row M = output neuron
            int k0   = ((2 * v) & 7) + 8 * ehi + 16 * (v >> 2);  // A-layout K for (lane,v)
            const float* W2 = (mlp == 0 ? sW2 : tW2) + l * 1024;
            HPack p;
            p.h[0] = (_Float16)W2[k0 * 32 + nOut];
            p.h[1] = (_Float16)W2[(k0 + 1) * 32 + nOut];
            aPack[mlp][tile][el * 8 + v] = p.u;
        }
        if (tid < 32) {
            pk1[0][tid] = make_float2(sW1[l * 64 + a * 32 + tid], sb1[l * 32 + tid]);
            pk1[1][tid] = make_float2(tW1[l * 64 + a * 32 + tid], tb1[l * 32 + tid]);
            b2l[0][tid] = sb2[l * 32 + tid];
            b2l[1][tid] = tb2[l * 32 + tid];
            pw3[0][tid] = make_float2(sW3[l * 64 + 2 * tid], sW3[l * 64 + 2 * tid + 1]);
            pw3[1][tid] = make_float2(tW3[l * 64 + 2 * tid], tW3[l * 64 + 2 * tid + 1]);
            if (tid < 2) {
                b3l[0][tid] = sb3[l * 2 + tid];
                b3l[1][tid] = tb3[l * 2 + tid];
                scl[tid]    = scale[l * 2 + tid];
            }
        }
        __syncthreads();

        // per-wave A registers (weights, reused across all sample tiles)
        AMat Areg[2][2];
        #pragma unroll
        for (int mlp = 0; mlp < 2; ++mlp)
            #pragma unroll
            for (int tile = 0; tile < 2; ++tile)
                #pragma unroll
                for (int v = 0; v < 8; ++v)
                    Areg[mlp][tile].u[v] = aPack[mlp][tile][lane * 8 + v];

        const float sc0 = scl[0], sc1 = scl[1];

        #pragma unroll
        for (int t = 0; t < TILES; ++t) {
            float xa = (a == 0) ? xs0[t] : xs1[t];
            float o[2][2];
            #pragma unroll
            for (int mlp = 0; mlp < 2; ++mlp) {
                // ---- layer 1 (rank-1: masked input has one nonzero coord) -> B (f16, B-layout)
                AMat B;
                #pragma unroll
                for (int j = 0; j < 16; ++j) {
                    int k = (j & 7) + 8 * hi + 16 * (j >> 3);   // B-layout: K = 8*hi + j (j<8), 16 + 8*hi + (j-8) (j>=8)
                    float2 wb = pk1[mlp][k];
                    float  h  = fmaf(xa, wb.x, wb.y);
                    B.e[j] = (_Float16)leaky(h);
                }
                // ---- layer 2: two WMMAs (output neurons 0..15 and 16..31), bias in C
                v8f clo, chi;
                #pragma unroll
                for (int r = 0; r < 8; ++r) {
                    clo[r] = b2l[mlp][r + 8 * hi];
                    chi[r] = b2l[mlp][16 + r + 8 * hi];
                }
                clo = __builtin_amdgcn_wmma_f32_16x16x32_f16(
                    false, Areg[mlp][0].v, false, B.v, (short)0, clo, false, false);
                chi = __builtin_amdgcn_wmma_f32_16x16x32_f16(
                    false, Areg[mlp][1].v, false, B.v, (short)0, chi, false, false);
                // ---- layer 3: per-lane 16-neuron dot + one partner exchange
                float p0 = 0.0f, p1 = 0.0f;
                #pragma unroll
                for (int r = 0; r < 8; ++r) {
                    float h2a = leaky(clo[r]);                 // neuron r + 8*hi
                    float h2b = leaky(chi[r]);                 // neuron 16 + r + 8*hi
                    float2 wA = pw3[mlp][r + 8 * hi];
                    float2 wB = pw3[mlp][16 + r + 8 * hi];
                    p0 = fmaf(h2a, wA.x, p0); p0 = fmaf(h2b, wB.x, p0);
                    p1 = fmaf(h2a, wA.y, p1); p1 = fmaf(h2b, wB.y, p1);
                }
                p0 += __shfl_xor(p0, 16, 32);
                p1 += __shfl_xor(p1, 16, 32);
                o[mlp][0] = p0 + b3l[mlp][0];
                o[mlp][1] = p1 + b3l[mlp][1];
            }
            // ---- coupling update: only inactive coordinate changes
            float s0 = o[0][0], s1 = o[0][1];
            float t0 = o[1][0], t1 = o[1][1];
            if (a == 0) {
                xs1[t] = fmaf(__expf(s1 * sc1), xs1[t], t1);
            } else {
                xs0[t] = fmaf(__expf(s0 * sc0), xs0[t], t0);
            }
            ldj += valid[t] ? (s0 + s1) : 0.0f;   // duplicated across partner lanes -> halved later
        }
    }

    // ---- per-sample log N(y | 0, I2)
    const float c0 = -1.8378770664093453f;   // -log(2*pi)
    #pragma unroll
    for (int t = 0; t < TILES; ++t) {
        int n = base + t * 16 + m;
        if (hi == 0 && n < N)
            out[n] = c0 - 0.5f * (xs0[t] * xs0[t] + xs1[t] * xs1[t]);
    }

    // ---- deterministic block-level ldj partial
    #pragma unroll
    for (int off = 16; off > 0; off >>= 1) ldj += __shfl_xor(ldj, off, 32);
    if (lane == 0) redbuf[wave] = ldj;
    __syncthreads();
    if (tid == 0) {
        float s = 0.0f;
        for (int w = 0; w < WAVES; ++w) s += redbuf[w];
        partials[blockIdx.x] = 0.5f * s;   // halve: samples duplicated across lane halves
    }
}

// ---------------------------------------------------------------------------
// Kernel 2: deterministic reduction of block partials -> single ldj scalar
// ---------------------------------------------------------------------------
__global__ __launch_bounds__(256) void reduce_kernel(const float* __restrict__ partials,
                                                     int nPart, float* __restrict__ outScalar)
{
    __shared__ float buf[256];
    float s = 0.0f;
    for (int i = threadIdx.x; i < nPart; i += 256) s += partials[i];
    buf[threadIdx.x] = s;
    __syncthreads();
    for (int off = 128; off > 0; off >>= 1) {
        if (threadIdx.x < off) buf[threadIdx.x] += buf[threadIdx.x + off];
        __syncthreads();
    }
    if (threadIdx.x == 0) *outScalar = buf[0];
}

// ---------------------------------------------------------------------------
// Kernel 3: out[n] += ldj  (log_p_y + log_det_jac, ldj is a global scalar)
// ---------------------------------------------------------------------------
__global__ __launch_bounds__(256) void add_scalar_kernel(float* __restrict__ out,
                                                         const float* __restrict__ sc, int n)
{
    int i = blockIdx.x * 256 + threadIdx.x;
    if (i < n) out[i] += *sc;
}

extern "C" void kernel_launch(void* const* d_in, const int* in_sizes, int n_in,
                              void* d_out, int out_size, void* d_ws, size_t ws_size,
                              hipStream_t stream)
{
    (void)n_in; (void)ws_size;
    const float* X    = (const float*)d_in[0];
    const float* sW1v = (const float*)d_in[1];
    const float* sb1v = (const float*)d_in[2];
    const float* sW2v = (const float*)d_in[3];
    const float* sb2v = (const float*)d_in[4];
    const float* sW3v = (const float*)d_in[5];
    const float* sb3v = (const float*)d_in[6];
    const float* tW1v = (const float*)d_in[7];
    const float* tb1v = (const float*)d_in[8];
    const float* tW2v = (const float*)d_in[9];
    const float* tb2v = (const float*)d_in[10];
    const float* tW3v = (const float*)d_in[11];
    const float* tb3v = (const float*)d_in[12];
    const float* scv  = (const float*)d_in[13];
    const float* mkv  = (const float*)d_in[14];

    const int N = in_sizes[0] / 2;
    const int blocks = (N + SAMPLES_PER_BLOCK - 1) / SAMPLES_PER_BLOCK;

    float* partials = (float*)d_ws;
    float* scalar   = partials + blocks;

    flow_kernel<<<blocks, BLOCK, 0, stream>>>(
        X, sW1v, sb1v, sW2v, sb2v, sW3v, sb3v,
        tW1v, tb1v, tW2v, tb2v, tW3v, tb3v,
        scv, mkv, (float*)d_out, partials, N);

    reduce_kernel<<<1, 256, 0, stream>>>(partials, blocks, scalar);

    add_scalar_kernel<<<(out_size + 255) / 256, 256, 0, stream>>>(
        (float*)d_out, scalar, out_size);
}
